// DGCNNMultiModalCondT_23905787970105
// MI455X (gfx1250) — matrix operand order repaired and checked by reading, not compiled, not run on hardware
//
#include <hip/hip_runtime.h>
#include <hip/hip_bf16.h>

typedef __attribute__((ext_vector_type(16))) _Float16 v16h;
typedef __attribute__((ext_vector_type(8)))  float    v8f;

#define FRAMES 128   // B*T
#define NPTS   256
#define KNN    20

// ---------------------------------------------------------------------------
// WMMA helpers (CDNA5 wave32, V_WMMA_F32_16X16X32_F16)
// ---------------------------------------------------------------------------
__device__ __forceinline__ v8f wmma16(v16h a, v16h b, v8f c) {
  return __builtin_amdgcn_wmma_f32_16x16x32_f16(false, a, false, b, (short)0, c,
                                                false, false);
}

// A fragment (16x32, M x K), source row-major [rows x lda] in LDS, no bounds.
__device__ __forceinline__ v16h frag_a_lds(const _Float16* A, int lda, int m0,
                                           int k0, int lane) {
  v16h a;
  int m = m0 + (lane & 15);
  int hi = lane >> 4;
#pragma unroll
  for (int e = 0; e < 16; ++e) {
    int k = k0 + ((e >> 3) << 4) + (hi << 3) + (e & 7);
    a[e] = A[m * lda + k];
  }
  return a;
}

// B fragment (32x16, K x N) where logical B[k][n] = W[n][k], W row-major NxK (LDS).
__device__ __forceinline__ v16h frag_b_lds(const _Float16* W, int ldw, int n0,
                                           int k0, int lane) {
  v16h b;
  int n = n0 + (lane & 15);
  int hi = lane >> 4;
#pragma unroll
  for (int e = 0; e < 16; ++e) {
    int k = k0 + ((e >> 3) << 4) + (hi << 3) + (e & 7);
    b[e] = W[n * ldw + k];
  }
  return b;
}

// Global, bounds-guarded variants (slow path, used only on partial M tiles).
__device__ __forceinline__ v16h frag_a_g(const _Float16* A, int M, int Kd,
                                         int m0, int k0, int lane) {
  v16h a;
  int m = m0 + (lane & 15);
  int hi = lane >> 4;
#pragma unroll
  for (int e = 0; e < 16; ++e) {
    int k = k0 + ((e >> 3) << 4) + (hi << 3) + (e & 7);
    _Float16 v = (_Float16)0.f;
    if (m < M && k < Kd) v = A[(long)m * Kd + k];
    a[e] = v;
  }
  return a;
}

// Fast vectorized global fragment loads.
// Layout facts (wave32, 16x16x32 f16):
//   A lane (m = m0+lane&15) needs halves [k0+8*hi .. +8) and [k0+16+8*hi .. +8)
//   B lane (n = n0+lane&15) needs halves [k0+8*hi .. +8) and [k0+16+8*hi .. +8) of W row n
// Valid when the tile is fully in-bounds and Kd % 32 == 0 (true for all calls).
union frag_u {
  uint4 q[2];
  v16h h;
};

__device__ __forceinline__ v16h frag_a_g_fast(const _Float16* A, int Kd, int m0,
                                              int k0, int lane) {
  int m = m0 + (lane & 15);
  int hi = lane >> 4;
  const _Float16* p = A + (long)m * Kd + k0 + hi * 8;
  frag_u u;
  u.q[0] = *(const uint4*)p;
  u.q[1] = *(const uint4*)(p + 16);
  return u.h;
}

__device__ __forceinline__ v16h frag_b_g_fast(const _Float16* W, int Kd, int n0,
                                              int k0, int lane) {
  int n = n0 + (lane & 15);
  int hi = lane >> 4;
  const _Float16* p = W + (long)n * Kd + k0 + hi * 8;
  frag_u u;
  u.q[0] = *(const uint4*)p;
  u.q[1] = *(const uint4*)(p + 16);
  return u.h;
}

__device__ __forceinline__ float sigmoidf_(float x) {
  return 1.f / (1.f + expf(-x));
}

// ---------------------------------------------------------------------------
// f32 -> f16 convert (weights / activations)
// ---------------------------------------------------------------------------
__global__ void f32_to_f16_kernel(const float* __restrict__ src,
                                  _Float16* __restrict__ dst, long n) {
  long i = (long)blockIdx.x * 256 + threadIdx.x;
  long stride = (long)gridDim.x * 256;
  for (; i < n; i += stride) dst[i] = (_Float16)src[i];
}

// ---------------------------------------------------------------------------
// prep: split point_cloud into geom/aux, padded f16 geom, squared norms
// ---------------------------------------------------------------------------
__global__ void prep0_kernel(const float* __restrict__ pc, float* geom32,
                             float* aux32, _Float16* geom16, float* sqn) {
  int n = blockIdx.x * 256 + threadIdx.x;
  if (n >= FRAMES * NPTS) return;
  const float* p = pc + (long)n * 6;
  float s = 0.f;
  for (int d = 0; d < 3; ++d) {
    float g = p[d];
    geom32[n * 3 + d] = g;
    aux32[n * 3 + d] = p[3 + d];
    geom16[(long)n * 32 + d] = (_Float16)g;
    s += g * g;
  }
  for (int d = 3; d < 32; ++d) geom16[(long)n * 32 + d] = (_Float16)0.f;
  sqn[n] = s;
}

// ---------------------------------------------------------------------------
// kNN Gram matrix per frame via WMMA: d2 = |xi|^2 + |xj|^2 - 2 xi.xj (+diag 1e9)
// grid = FRAMES, block = 256 (8 waves)
// ---------------------------------------------------------------------------
__global__ void gram_kernel(const _Float16* __restrict__ xh,
                            const float* __restrict__ sqn,
                            float* __restrict__ d2) {
  __shared__ _Float16 xl[NPTS * 32];
  int f = blockIdx.x, tid = threadIdx.x;
  {
    const uint4* s4 = (const uint4*)(xh + (long)f * NPTS * 32);
    uint4* dl = (uint4*)xl;
    for (int t = tid; t < NPTS * 32 * 2 / 16; t += 256) dl[t] = s4[t];
  }
  __syncthreads();
  int wave = tid >> 5, lane = tid & 31;
  const float* sq = sqn + f * NPTS;
  float* out = d2 + (long)f * NPTS * NPTS;
  for (int t = wave; t < 256; t += 8) {
    int tm = t >> 4, tn = t & 15;
    v8f c = {};
    v16h a = frag_a_lds(xl, 32, tm * 16, 0, lane);
    v16h b = frag_b_lds(xl, 32, tn * 16, 0, lane);
    c = wmma16(a, b, c);
    int hi = lane >> 4, col = tn * 16 + (lane & 15);
    float sqc = sq[col];
#pragma unroll
    for (int r = 0; r < 8; ++r) {
      int row = tm * 16 + r + 8 * hi;
      float v = sq[row] + sqc - 2.f * c[r];
      if (row == col) v += 1e9f;
      out[row * NPTS + col] = v;
    }
  }
}

// ---------------------------------------------------------------------------
// per-row top-K (smallest distances) -> neighbor indices
// ---------------------------------------------------------------------------
__global__ void topk_kernel(const float* __restrict__ d2, int* __restrict__ idx) {
  int row = blockIdx.x * 256 + threadIdx.x;
  if (row >= FRAMES * NPTS) return;
  const float* dr = d2 + (long)row * NPTS;
  float bd[KNN];
  int bi[KNN];
#pragma unroll
  for (int t = 0; t < KNN; ++t) { bd[t] = 3e38f; bi[t] = 0; }
  for (int j = 0; j < NPTS; ++j) {
    float d = dr[j];
    if (d < bd[KNN - 1]) {
      int p = KNN - 1;
      while (p > 0 && bd[p - 1] > d) {
        bd[p] = bd[p - 1]; bi[p] = bi[p - 1]; --p;
      }
      bd[p] = d; bi[p] = j;
    }
  }
#pragma unroll
  for (int t = 0; t < KNN; ++t) idx[(long)row * KNN + t] = bi[t];
}

// ---------------------------------------------------------------------------
// Fused EdgeConv: gather -> eMLP (2 WMMA GEMMs, BN folded) -> auxMLP (2 WMMA
// GEMMs) -> FiLM gate -> max over K -> LayerNorm -> relu. Also emits f16 copy
// and squared norms for the next layer's kNN.
// grid = FRAMES * (NPTS/8), block = 256. One workgroup = 8 centers (160 edges).
// ---------------------------------------------------------------------------
__global__ void edge_kernel(const float* __restrict__ xsrc, int xstride, int Din,
                            const float* __restrict__ aux,
                            const int* __restrict__ idx,
                            const float* e_w0, const float* e_b0,
                            const float* bn0_g, const float* bn0_b,
                            const float* bn0_m, const float* bn0_v,
                            const float* e_w1, const float* e_b1,
                            const float* bn1_g, const float* bn1_b,
                            const float* bn1_m, const float* bn1_v,
                            const float* a_w0, const float* a_b0,
                            const float* a_w1, const float* a_b1,
                            const float* ln_g, const float* ln_b,
                            float* __restrict__ xout,
                            _Float16* __restrict__ xout16,
                            float* __restrict__ sqn) {
  extern __shared__ char smem[];
  _Float16* eg    = (_Float16*)smem;        // 160 x 64
  _Float16* aux16 = eg    + 160 * 64;       // 160 x 32
  _Float16* h16   = aux16 + 160 * 32;       // 160 x 32
  _Float16* g116  = h16   + 160 * 32;       // 160 x 64
  _Float16* we0   = g116  + 160 * 64;       // 32 x 64
  _Float16* we1   = we0   + 32 * 64;        // 32 x 32
  _Float16* wa0   = we1   + 32 * 32;        // 64 x 32
  _Float16* wa1   = wa0   + 64 * 32;        // 64 x 64
  float* ef32 = (float*)(wa1 + 64 * 64);    // 160 x 32
  float* gb32 = ef32 + 160 * 32;            // 160 x 64
  float* sc0 = gb32 + 160 * 64;             // 32
  float* of0 = sc0 + 32;
  float* sc1 = of0 + 32;
  float* of1 = sc1 + 32;
  float* ab0 = of1 + 32;                    // 64
  float* ab1 = ab0 + 64;                    // 64
  float* lng = ab1 + 64;                    // 32
  float* lnb = lng + 32;                    // 32

  int f = blockIdx.x >> 5;
  int i0 = (blockIdx.x & 31) * 8;
  int tid = threadIdx.x;
  int twoD = 2 * Din;

  for (int t = tid; t < 32 * 64; t += 256) {
    int n = t >> 6, k = t & 63;
    we0[t] = (_Float16)((k < twoD) ? e_w0[n * twoD + k] : 0.f);
  }
  for (int t = tid; t < 32 * 32; t += 256) we1[t] = (_Float16)e_w1[t];
  for (int t = tid; t < 64 * 32; t += 256) {
    int n = t >> 5, k = t & 31;
    wa0[t] = (_Float16)((k < 6) ? a_w0[n * 6 + k] : 0.f);
  }
  for (int t = tid; t < 64 * 64; t += 256) wa1[t] = (_Float16)a_w1[t];
  if (tid < 32) {
    float s = bn0_g[tid] * rsqrtf(bn0_v[tid] + 1e-5f);
    sc0[tid] = s;
    of0[tid] = (e_b0[tid] - bn0_m[tid]) * s + bn0_b[tid];
    float s1 = bn1_g[tid] * rsqrtf(bn1_v[tid] + 1e-5f);
    sc1[tid] = s1;
    of1[tid] = (e_b1[tid] - bn1_m[tid]) * s1 + bn1_b[tid];
    lng[tid] = ln_g[tid];
    lnb[tid] = ln_b[tid];
  }
  if (tid < 64) { ab0[tid] = a_b0[tid]; ab1[tid] = a_b1[tid]; }

  if (tid < 160) {
    int c = tid / KNN, kk = tid % KNN;
    int ic = i0 + c;
    int j = idx[((long)f * NPTS + ic) * KNN + kk];
    const float* xi = xsrc + ((long)f * NPTS + ic) * xstride;
    const float* xj = xsrc + ((long)f * NPTS + j) * xstride;
    for (int d = 0; d < 64; ++d) {
      float v = 0.f;
      if (d < Din) v = xi[d];
      else if (d < twoD) v = xj[d - Din] - xi[d - Din];
      eg[tid * 64 + d] = (_Float16)v;
    }
    const float* ai = aux + ((long)f * NPTS + ic) * 3;
    const float* aj = aux + ((long)f * NPTS + j) * 3;
    for (int d = 0; d < 32; ++d) {
      float v = 0.f;
      if (d < 3) v = ai[d]; else if (d < 6) v = aj[d - 3];
      aux16[tid * 32 + d] = (_Float16)v;
    }
  }
  __syncthreads();

  int wave = tid >> 5, lane = tid & 31;
  int hi = lane >> 4;

  // stage 1: h = relu(bn0(eg @ e_w0^T + b0))   M=160 N=32 K=64
  for (int t = wave; t < 20; t += 8) {
    int tm = t >> 1, tn = t & 1;
    v8f c = {};
    for (int ks = 0; ks < 2; ++ks) {
      v16h a = frag_a_lds(eg, 64, tm * 16, ks * 32, lane);
      v16h b = frag_b_lds(we0, 64, tn * 16, ks * 32, lane);
      c = wmma16(a, b, c);
    }
    int col = tn * 16 + (lane & 15);
    float s = sc0[col], o = of0[col];
#pragma unroll
    for (int r = 0; r < 8; ++r) {
      int row = tm * 16 + r + 8 * hi;
      float y = c[r] * s + o;
      y = y > 0.f ? y : 0.f;
      h16[row * 32 + col] = (_Float16)y;
    }
  }
  __syncthreads();

  // stage 2: ef = relu(bn1(h @ e_w1^T + b1))   M=160 N=32 K=32
  for (int t = wave; t < 20; t += 8) {
    int tm = t >> 1, tn = t & 1;
    v8f c = {};
    v16h a = frag_a_lds(h16, 32, tm * 16, 0, lane);
    v16h b = frag_b_lds(we1, 32, tn * 16, 0, lane);
    c = wmma16(a, b, c);
    int col = tn * 16 + (lane & 15);
    float s = sc1[col], o = of1[col];
#pragma unroll
    for (int r = 0; r < 8; ++r) {
      int row = tm * 16 + r + 8 * hi;
      float y = c[r] * s + o;
      y = y > 0.f ? y : 0.f;
      ef32[row * 32 + col] = y;
    }
  }
  // stage 3: g1 = relu(aux @ a_w0^T + a_b0)    M=160 N=64 K=32
  for (int t = wave; t < 40; t += 8) {
    int tm = t >> 2, tn = t & 3;
    v8f c = {};
    v16h a = frag_a_lds(aux16, 32, tm * 16, 0, lane);
    v16h b = frag_b_lds(wa0, 32, tn * 16, 0, lane);
    c = wmma16(a, b, c);
    int col = tn * 16 + (lane & 15);
    float o = ab0[col];
#pragma unroll
    for (int r = 0; r < 8; ++r) {
      int row = tm * 16 + r + 8 * hi;
      float y = c[r] + o;
      y = y > 0.f ? y : 0.f;
      g116[row * 64 + col] = (_Float16)y;
    }
  }
  __syncthreads();

  // stage 4: gb = g1 @ a_w1^T + a_b1           M=160 N=64 K=64
  for (int t = wave; t < 40; t += 8) {
    int tm = t >> 2, tn = t & 3;
    v8f c = {};
    for (int ks = 0; ks < 2; ++ks) {
      v16h a = frag_a_lds(g116, 64, tm * 16, ks * 32, lane);
      v16h b = frag_b_lds(wa1, 64, tn * 16, ks * 32, lane);
      c = wmma16(a, b, c);
    }
    int col = tn * 16 + (lane & 15);
    float o = ab1[col];
#pragma unroll
    for (int r = 0; r < 8; ++r) {
      int row = tm * 16 + r + 8 * hi;
      gb32[row * 64 + col] = c[r] + o;
    }
  }
  __syncthreads();

  // stage 5: FiLM gate + max over K + LayerNorm + relu (one wave per center)
  {
    int c8 = tid >> 5, ch = tid & 31;
    float mx = -1e30f;
    for (int kk = 0; kk < KNN; ++kk) {
      int e = c8 * KNN + kk;
      float gate = sigmoidf_(gb32[e * 64 + ch] + 1.f);
      float val = gate * ef32[e * 32 + ch] + gb32[e * 64 + 32 + ch];
      mx = fmaxf(mx, val);
    }
    float sum = mx;
    for (int o = 16; o > 0; o >>= 1) sum += __shfl_xor(sum, o, 32);
    float mu = sum * (1.f / 32.f);
    float d = mx - mu;
    float vs = d * d;
    for (int o = 16; o > 0; o >>= 1) vs += __shfl_xor(vs, o, 32);
    float rs = rsqrtf(vs * (1.f / 32.f) + 1e-5f);
    float y = d * rs * lng[ch] + lnb[ch];
    y = y > 0.f ? y : 0.f;
    long node = (long)f * NPTS + i0 + c8;
    xout[node * 32 + ch] = y;
    xout16[node * 32 + ch] = (_Float16)y;
    float ss = y * y;
    for (int o = 16; o > 0; o >>= 1) ss += __shfl_xor(ss, o, 32);
    if (ch == 0) sqn[node] = ss;
  }
}

// ---------------------------------------------------------------------------
// concat the three 32-ch feature maps into xc16 (F*N x 96, f16)
// ---------------------------------------------------------------------------
__global__ void concat_kernel(const float* x1, const float* x2, const float* x3,
                              _Float16* xc) {
  int n = blockIdx.x * 256 + threadIdx.x;
  if (n >= FRAMES * NPTS) return;
  for (int d = 0; d < 32; ++d) {
    xc[(long)n * 96 + d]      = (_Float16)x1[(long)n * 32 + d];
    xc[(long)n * 96 + 32 + d] = (_Float16)x2[(long)n * 32 + d];
    xc[(long)n * 96 + 64 + d] = (_Float16)x3[(long)n * 32 + d];
  }
}

// ---------------------------------------------------------------------------
// lin1 (96->1024) + BN + relu + global max pool over N, fused WMMA kernel.
// grid = FRAMES * 8 (128 cols per block), block = 256.
// ---------------------------------------------------------------------------
__global__ void lin1max_kernel(const _Float16* __restrict__ xc,
                               const _Float16* __restrict__ w16,
                               const float* bias, const float* bn_g,
                               const float* bn_b, const float* bn_m,
                               const float* bn_v, float* __restrict__ E) {
  extern __shared__ char smem[];
  _Float16* xl = (_Float16*)smem;       // 256 x 96
  _Float16* wl = xl + 256 * 96;         // 128 x 96
  float* sc = (float*)(wl + 128 * 96);  // 128
  float* of = sc + 128;                 // 128
  int f = blockIdx.x >> 3;
  int nb = (blockIdx.x & 7) * 128;
  int tid = threadIdx.x;
  {
    const uint4* s = (const uint4*)(xc + (long)f * 256 * 96);
    uint4* d = (uint4*)xl;
    for (int t = tid; t < 3072; t += 256) d[t] = s[t];
  }
  {
    const uint4* s = (const uint4*)(w16 + (long)nb * 96);
    uint4* d = (uint4*)wl;
    for (int t = tid; t < 1536; t += 256) d[t] = s[t];
  }
  if (tid < 128) {
    int col = nb + tid;
    float s = bn_g[col] * rsqrtf(bn_v[col] + 1e-5f);
    sc[tid] = s;
    of[tid] = (bias[col] - bn_m[col]) * s + bn_b[col];
  }
  __syncthreads();
  int wave = tid >> 5, lane = tid & 31;
  int n0 = wave * 16;
  float s = sc[n0 + (lane & 15)], o = of[n0 + (lane & 15)];
  float cmax = -1e30f;
  for (int mt = 0; mt < 16; ++mt) {
    v8f c = {};
    for (int ks = 0; ks < 3; ++ks) {
      v16h a = frag_a_lds(xl, 96, mt * 16, ks * 32, lane);
      v16h b = frag_b_lds(wl, 96, n0, ks * 32, lane);
      c = wmma16(a, b, c);
    }
#pragma unroll
    for (int r = 0; r < 8; ++r) {
      float y = c[r] * s + o;
      y = y > 0.f ? y : 0.f;
      cmax = fmaxf(cmax, y);
    }
  }
  cmax = fmaxf(cmax, __shfl_xor(cmax, 16, 32));
  if (lane < 16) E[(long)f * 1024 + nb + n0 + lane] = cmax;
}

// ---------------------------------------------------------------------------
// cross-modal gating + positional embedding (VALU). grid = FRAMES, block = 256
// ---------------------------------------------------------------------------
__global__ void gating_kernel(const float* __restrict__ fs, const float* mn_g,
                              const float* mn_b, const float* wq,
                              const float* wk, const float* wv, const float* wo,
                              const float* pos, float* __restrict__ E,
                              _Float16* __restrict__ E16) {
  __shared__ float Er[1024];
  __shared__ float s9[9];
  __shared__ float gate[64];
  int f = blockIdx.x, tid = threadIdx.x;
  float* Eg = E + (long)f * 1024;
  for (int t = tid; t < 1024; t += 256) Er[t] = Eg[t];
  if (tid < 3) {
    const float* c = fs + f * 9 + tid * 3;
    float mu = (c[0] + c[1] + c[2]) * (1.f / 3.f);
    float v = ((c[0] - mu) * (c[0] - mu) + (c[1] - mu) * (c[1] - mu) +
               (c[2] - mu) * (c[2] - mu)) * (1.f / 3.f);
    float rs = rsqrtf(v + 1e-5f);
    for (int j = 0; j < 3; ++j)
      s9[tid * 3 + j] = (c[j] - mu) * rs * mn_g[tid * 3 + j] + mn_b[tid * 3 + j];
  }
  __syncthreads();
  if (tid < 64) {
    float q = 0.f;
    const float* wr = wq + (long)tid * 1024;
    for (int d = 0; d < 1024; ++d) q += Er[d] * wr[d];
    float kk = 0.f, vv = 0.f;
    for (int d = 0; d < 9; ++d) {
      kk += s9[d] * wk[tid * 9 + d];
      vv += s9[d] * wv[tid * 9 + d];
    }
    gate[tid] = sigmoidf_(q * kk * 0.125f) * vv;
  }
  __syncthreads();
  int t32 = f & 31;
  for (int d = tid; d < 1024; d += 256) {
    float acc = 0.f;
    for (int j = 0; j < 64; ++j) acc += gate[j] * wo[(long)d * 64 + j];
    float y = Er[d] + acc + pos[(long)t32 * 1024 + d];
    Eg[d] = y;
    E16[(long)f * 1024 + d] = (_Float16)y;
  }
}

// ---------------------------------------------------------------------------
// generic WMMA GEMM: C = A(MxK) @ W^T(NxK) [+bias][+resid][relu], opt f16 copy
// grid = (ceil(M/16), ceil(N/64)), block = (32,4); one wave per 16x16 tile.
// Requires Kd % 32 == 0 and N % 16 == 0 (true for all call sites).
// ---------------------------------------------------------------------------
__global__ void gemm_kernel(const _Float16* __restrict__ A,
                            const _Float16* __restrict__ W, const float* bias,
                            const float* resid, float* __restrict__ C,
                            _Float16* __restrict__ C16, int M, int N, int Kd,
                            int relu) {
  int lane = threadIdx.x;
  int nt = blockIdx.y * 4 + threadIdx.y;
  if (nt * 16 >= N) return;  // uniform per wave
  int m0 = blockIdx.x * 16;
  int n0 = nt * 16;
  v8f c = {};
  int ks = Kd >> 5;
  bool fullM = (m0 + 16 <= M);  // uniform per wave
  if (fullM) {
    for (int k = 0; k < ks; ++k) {
      if (k + 1 < ks)
        __builtin_prefetch(W + (long)(n0 + (lane & 15)) * Kd + (k + 1) * 32, 0, 3);
      v16h a = frag_a_g_fast(A, Kd, m0, k * 32, lane);
      v16h b = frag_b_g_fast(W, Kd, n0, k * 32, lane);
      c = wmma16(a, b, c);
    }
  } else {
    for (int k = 0; k < ks; ++k) {
      v16h a = frag_a_g(A, M, Kd, m0, k * 32, lane);
      v16h b = frag_b_g_fast(W, Kd, n0, k * 32, lane);
      c = wmma16(a, b, c);
    }
  }
  int hi = lane >> 4, col = n0 + (lane & 15);
  float bz = bias ? bias[col] : 0.f;
#pragma unroll
  for (int r = 0; r < 8; ++r) {
    int row = m0 + r + 8 * hi;
    if (row < M) {
      float y = c[r] + bz;
      if (resid) y += resid[(long)row * N + col];
      if (relu) y = y > 0.f ? y : 0.f;
      C[(long)row * N + col] = y;
      if (C16) C16[(long)row * N + col] = (_Float16)y;
    }
  }
}

// ---------------------------------------------------------------------------
// temporal self-attention (B=4, H=4, T=32, dh=256). grid = 16, block = 256
// ---------------------------------------------------------------------------
__global__ void attn_kernel(const float* __restrict__ qkv,
                            float* __restrict__ o32,
                            _Float16* __restrict__ o16) {
  extern __shared__ char smem[];
  float* qS = (float*)smem;       // 32 x 256
  float* kS = qS + 32 * 256;
  float* vS = kS + 32 * 256;
  float* sc = vS + 32 * 256;      // 32 x 32
  int bh = blockIdx.x;
  int b = bh >> 2, h = bh & 3;
  int tid = threadIdx.x;
  for (int t = tid; t < 32 * 256; t += 256) {
    int tt = t >> 8, d = t & 255;
    long row = (long)(b * 32 + tt) * 3072;
    qS[t] = qkv[row + h * 256 + d];
    kS[t] = qkv[row + 1024 + h * 256 + d];
    vS[t] = qkv[row + 2048 + h * 256 + d];
  }
  __syncthreads();
  for (int t = tid; t < 1024; t += 256) {
    int tt = t >> 5, ss = t & 31;
    float acc = 0.f;
    for (int d = 0; d < 256; ++d) acc += qS[tt * 256 + d] * kS[ss * 256 + d];
    sc[t] = acc * (1.f / 16.f);
  }
  __syncthreads();
  if (tid < 32) {
    float mx = -1e30f;
    for (int ss = 0; ss < 32; ++ss) mx = fmaxf(mx, sc[tid * 32 + ss]);
    float sum = 0.f;
    for (int ss = 0; ss < 32; ++ss) {
      float e = expf(sc[tid * 32 + ss] - mx);
      sc[tid * 32 + ss] = e;
      sum += e;
    }
    float inv = 1.f / sum;
    for (int ss = 0; ss < 32; ++ss) sc[tid * 32 + ss] *= inv;
  }
  __syncthreads();
  for (int t = tid; t < 32 * 256; t += 256) {
    int tt = t >> 8, d = t & 255;
    float acc = 0.f;
    for (int ss = 0; ss < 32; ++ss) acc += sc[tt * 32 + ss] * vS[ss * 256 + d];
    long oi = (long)(b * 32 + tt) * 1024 + h * 256 + d;
    o32[oi] = acc;
    o16[oi] = (_Float16)acc;
  }
}

// ---------------------------------------------------------------------------
// LayerNorm over 1024 per row. grid = 128, block = 256
// ---------------------------------------------------------------------------
__global__ void ln_kernel(const float* __restrict__ X, const float* g,
                          const float* b, float* __restrict__ out,
                          _Float16* __restrict__ out16) {
  __shared__ float red[256];
  __shared__ float stats[2];
  int row = blockIdx.x, tid = threadIdx.x;
  const float* xr = X + (long)row * 1024;
  float s = 0.f;
  for (int i = tid; i < 1024; i += 256) s += xr[i];
  red[tid] = s;
  __syncthreads();
  for (int o = 128; o > 0; o >>= 1) {
    if (tid < o) red[tid] += red[tid + o];
    __syncthreads();
  }
  if (tid == 0) stats[0] = red[0] * (1.f / 1024.f);
  __syncthreads();
  float mu = stats[0];
  float v = 0.f;
  for (int i = tid; i < 1024; i += 256) {
    float d = xr[i] - mu;
    v += d * d;
  }
  __syncthreads();
  red[tid] = v;
  __syncthreads();
  for (int o = 128; o > 0; o >>= 1) {
    if (tid < o) red[tid] += red[tid + o];
    __syncthreads();
  }
  if (tid == 0) stats[1] = rsqrtf(red[0] * (1.f / 1024.f) + 1e-5f);
  __syncthreads();
  float rs = stats[1];
  for (int i = tid; i < 1024; i += 256) {
    float y = (xr[i] - mu) * rs * g[i] + b[i];
    out[(long)row * 1024 + i] = y;
    out16[(long)row * 1024 + i] = (_Float16)y;
  }
}

// mean over T -> (4,1024). grid = (4,4), block = 256
__global__ void mean_kernel(const float* __restrict__ E, float* g32,
                            _Float16* g16) {
  int b = blockIdx.x;
  int d = blockIdx.y * 256 + threadIdx.x;
  float acc = 0.f;
  for (int t = 0; t < 32; ++t) acc += E[(long)(b * 32 + t) * 1024 + d];
  acc *= (1.f / 32.f);
  g32[b * 1024 + d] = acc;
  g16[b * 1024 + d] = (_Float16)acc;
}

// ---------------------------------------------------------------------------
// Host side
// ---------------------------------------------------------------------------
enum {
  IN_PC = 0, IN_FS = 1,
  CA_MN_B = 2, CA_MN_G = 3, CA_WK = 4, CA_WO = 5, CA_WQ = 6, CA_WV = 7,
  EDGE0 = 8, EDGE1 = 26, EDGE2 = 44,
  // edge offsets: 0 a_b0, 1 a_b1, 2 a_w0, 3 a_w1, 4 e_b0, 5 e_b1,
  // 6 bn0_b, 7 bn0_g, 8 bn0_m, 9 bn0_v, 10 bn1_b, 11 bn1_g, 12 bn1_m,
  // 13 bn1_v, 14 e_w0, 15 e_w1, 16 ln_b, 17 ln_g
  HEAD0_W = 62, HEAD0_B = 63, HEAD1_W = 64, HEAD1_B = 65,
  HEAD2_W = 66, HEAD2_B = 67, HEAD3_W = 68, HEAD3_B = 69,
  LIN1_B = 70, LIN1_BN_B = 71, LIN1_BN_G = 72, LIN1_BN_M = 73, LIN1_BN_V = 74,
  LIN1_W = 75, POS = 76,
  TR_FF_B1 = 77, TR_FF_B2 = 78, TR_FF_W1 = 79, TR_FF_W2 = 80,
  TR_IN_B = 81, TR_IN_W = 82, TR_LN1_B = 83, TR_LN1_G = 84,
  TR_LN2_B = 85, TR_LN2_G = 86, TR_OUT_B = 87, TR_OUT_W = 88
};

extern "C" void kernel_launch(void* const* d_in, const int* in_sizes, int n_in,
                              void* d_out, int out_size, void* d_ws,
                              size_t ws_size, hipStream_t stream) {
  (void)in_sizes; (void)n_in; (void)out_size; (void)ws_size;
  const long FN = (long)FRAMES * NPTS;
  auto P = [&](int i) { return (const float*)d_in[i]; };

  // allow >64KB dynamic LDS for the big fused kernels (host-side attribute,
  // not a stream op -- safe under graph capture; errors ignored)
  const size_t EDGE_LDS = 142592;
  const size_t LIN1_LDS = 74752;
  const size_t ATTN_LDS = 102400;
  (void)hipFuncSetAttribute((const void*)edge_kernel,
                            hipFuncAttributeMaxDynamicSharedMemorySize,
                            (int)EDGE_LDS);
  (void)hipFuncSetAttribute((const void*)lin1max_kernel,
                            hipFuncAttributeMaxDynamicSharedMemorySize,
                            (int)LIN1_LDS);
  (void)hipFuncSetAttribute((const void*)attn_kernel,
                            hipFuncAttributeMaxDynamicSharedMemorySize,
                            (int)ATTN_LDS);

  char* base = (char*)d_ws;
  size_t off = 0;
  auto alloc = [&](size_t bytes) {
    size_t o = off;
    off += (bytes + 255) & ~(size_t)255;
    return (void*)(base + o);
  };
  float*    geom32  = (float*)alloc(FN * 3 * 4);
  float*    aux32   = (float*)alloc(FN * 3 * 4);
  _Float16* geom16  = (_Float16*)alloc(FN * 32 * 2);
  float*    sqn     = (float*)alloc(FN * 4);
  float*    d2      = (float*)alloc(FN * NPTS * 4);
  int*      idxb    = (int*)alloc(FN * KNN * 4);
  float*    X32[3];
  _Float16* XH[3];
  for (int l = 0; l < 3; ++l) {
    X32[l] = (float*)alloc(FN * 32 * 4);
    XH[l]  = (_Float16*)alloc(FN * 32 * 2);
  }
  _Float16* xc16   = (_Float16*)alloc(FN * 96 * 2);
  float*    E      = (float*)alloc(128 * 1024 * 4);
  _Float16* E16    = (_Float16*)alloc(128 * 1024 * 2);
  float*    Etmp   = (float*)alloc(128 * 1024 * 4);
  float*    qkv    = (float*)alloc(128 * 3072 * 4);
  float*    o32    = (float*)alloc(128 * 1024 * 4);
  _Float16* o16    = (_Float16*)alloc(128 * 1024 * 2);
  float*    ff1    = (float*)alloc(128 * 2048 * 4);
  _Float16* ff116  = (_Float16*)alloc(128 * 2048 * 2);
  float*    g32    = (float*)alloc(4 * 1024 * 4);
  _Float16* g16    = (_Float16*)alloc(4 * 1024 * 2);
  float*    h1     = (float*)alloc(4 * 1024 * 4);
  _Float16* h116   = (_Float16*)alloc(4 * 1024 * 2);
  float*    h2     = (float*)alloc(4 * 256 * 4);
  _Float16* h216   = (_Float16*)alloc(4 * 256 * 2);
  float*    h3     = (float*)alloc(4 * 128 * 4);
  _Float16* h316   = (_Float16*)alloc(4 * 128 * 2);
  _Float16* lin1w16 = (_Float16*)alloc((size_t)1024 * 96 * 2);
  _Float16* inw16   = (_Float16*)alloc((size_t)3072 * 1024 * 2);
  _Float16* outw16  = (_Float16*)alloc((size_t)1024 * 1024 * 2);
  _Float16* ffw116  = (_Float16*)alloc((size_t)2048 * 1024 * 2);
  _Float16* ffw216  = (_Float16*)alloc((size_t)1024 * 2048 * 2);
  _Float16* hw016   = (_Float16*)alloc((size_t)1024 * 1024 * 2);
  _Float16* hw116   = (_Float16*)alloc((size_t)256 * 1024 * 2);
  _Float16* hw216   = (_Float16*)alloc((size_t)128 * 256 * 2);
  _Float16* hw316   = (_Float16*)alloc((size_t)16 * 128 * 2);

  auto conv = [&](const float* s, _Float16* d, long n) {
    int blocks = (int)((n + 255) / 256);
    if (blocks > 8192) blocks = 8192;
    f32_to_f16_kernel<<<blocks, 256, 0, stream>>>(s, d, n);
  };

  // --- weight conversions (f32 -> f16, row-major preserved) ---
  conv(P(LIN1_W),   lin1w16, 1024L * 96);
  conv(P(TR_IN_W),  inw16,   3072L * 1024);
  conv(P(TR_OUT_W), outw16,  1024L * 1024);
  conv(P(TR_FF_W1), ffw116,  2048L * 1024);
  conv(P(TR_FF_W2), ffw216,  1024L * 2048);
  conv(P(HEAD0_W),  hw016,   1024L * 1024);
  conv(P(HEAD1_W),  hw116,   256L * 1024);
  conv(P(HEAD2_W),  hw216,   128L * 256);
  conv(P(HEAD3_W),  hw316,   16L * 128);

  // --- point cloud prep ---
  prep0_kernel<<<(FN + 255) / 256, 256, 0, stream>>>(P(IN_PC), geom32, aux32,
                                                     geom16, sqn);

  // --- 3 EdgeConv layers ---
  const int ebase[3] = {EDGE0, EDGE1, EDGE2};
  const int Din[3] = {3, 32, 32};
  const float* xsrc[3] = {geom32, X32[0], X32[1]};
  const int xstr[3] = {3, 32, 32};
  const _Float16* xhin[3] = {geom16, XH[0], XH[1]};
  for (int l = 0; l < 3; ++l) {
    gram_kernel<<<FRAMES, 256, 0, stream>>>(xhin[l], sqn, d2);
    topk_kernel<<<(FN + 255) / 256, 256, 0, stream>>>(d2, idxb);
    int eb = ebase[l];
    edge_kernel<<<FRAMES * (NPTS / 8), 256, EDGE_LDS, stream>>>(
        xsrc[l], xstr[l], Din[l], aux32, idxb,
        P(eb + 14), P(eb + 4), P(eb + 7), P(eb + 6), P(eb + 8), P(eb + 9),
        P(eb + 15), P(eb + 5), P(eb + 11), P(eb + 10), P(eb + 12), P(eb + 13),
        P(eb + 2), P(eb + 0), P(eb + 3), P(eb + 1), P(eb + 17), P(eb + 16),
        X32[l], XH[l], sqn);
  }

  // --- concat + lin1 + BN + relu + global max pool ---
  concat_kernel<<<(FN + 255) / 256, 256, 0, stream>>>(X32[0], X32[1], X32[2],
                                                      xc16);
  lin1max_kernel<<<FRAMES * 8, 256, LIN1_LDS, stream>>>(
      xc16, lin1w16, P(LIN1_B), P(LIN1_BN_G), P(LIN1_BN_B), P(LIN1_BN_M),
      P(LIN1_BN_V), E);

  // --- cross-modal gating + positional embedding ---
  gating_kernel<<<FRAMES, 256, 0, stream>>>(P(IN_FS), P(CA_MN_G), P(CA_MN_B),
                                            P(CA_WQ), P(CA_WK), P(CA_WV),
                                            P(CA_WO), P(POS), E, E16);

  // --- transformer ---
  gemm_kernel<<<dim3(8, 48), dim3(32, 4), 0, stream>>>(
      E16, inw16, P(TR_IN_B), nullptr, qkv, nullptr, 128, 3072, 1024, 0);
  attn_kernel<<<16, 256, ATTN_LDS, stream>>>(qkv, o32, o16);
  gemm_kernel<<<dim3(8, 16), dim3(32, 4), 0, stream>>>(
      o16, outw16, P(TR_OUT_B), E, Etmp, nullptr, 128, 1024, 1024, 0);
  ln_kernel<<<128, 256, 0, stream>>>(Etmp, P(TR_LN1_G), P(TR_LN1_B), E, E16);
  gemm_kernel<<<dim3(8, 32), dim3(32, 4), 0, stream>>>(
      E16, ffw116, P(TR_FF_B1), nullptr, ff1, ff116, 128, 2048, 1024, 1);
  gemm_kernel<<<dim3(8, 16), dim3(32, 4), 0, stream>>>(
      ff116, ffw216, P(TR_FF_B2), E, Etmp, nullptr, 128, 1024, 2048, 0);
  ln_kernel<<<128, 256, 0, stream>>>(Etmp, P(TR_LN2_G), P(TR_LN2_B), E, E16);

  // --- mean over T + classifier head ---
  mean_kernel<<<dim3(4, 4), 256, 0, stream>>>(E, g32, g16);
  gemm_kernel<<<dim3(1, 16), dim3(32, 4), 0, stream>>>(
      g16, hw016, P(HEAD0_B), nullptr, h1, h116, 4, 1024, 1024, 1);
  gemm_kernel<<<dim3(1, 4), dim3(32, 4), 0, stream>>>(
      h116, hw116, P(HEAD1_B), nullptr, h2, h216, 4, 256, 1024, 1);
  gemm_kernel<<<dim3(1, 2), dim3(32, 4), 0, stream>>>(
      h216, hw216, P(HEAD2_B), nullptr, h3, h316, 4, 128, 256, 1);
  gemm_kernel<<<dim3(1, 1), dim3(32, 4), 0, stream>>>(
      h316, hw316, P(HEAD3_B), nullptr, (float*)d_out, nullptr, 4, 16, 128, 0);
}
